// MADA_networks_85383949844807
// MI455X (gfx1250) — matrix operand order repaired and checked by reading, not compiled, not run on hardware
//
#include <hip/hip_runtime.h>
#include <hip/hip_bf16.h>
#include <stdint.h>

// ---------------------------------------------------------------------------
// MI455X (gfx1250). Matrix-op bound (~160 GFLOP vs ~110MB unique IO), so all
// GEMMs run on v_wmma_f32_16x16x32_bf16 with f32 accumulation.
// GEMM: 128x128 block tile, 4 waves, 64x64 per wave (16 WMMA : 16 ds_b128 per
// K-step), LDS double-buffered via GLOBAL_LOAD_ASYNC_TO_LDS_B128 (ASYNCcnt,
// no staging VGPRs), one barrier per K-step.
// ---------------------------------------------------------------------------

typedef __bf16 bf16_t;
typedef __attribute__((ext_vector_type(16))) __bf16 v16bf;
typedef __attribute__((ext_vector_type(8)))  __bf16 v8bf;
typedef __attribute__((ext_vector_type(8)))  float  v8f;
typedef __attribute__((ext_vector_type(4)))  int    v4i;

#define B_DIM   4096
#define DIN     2048
#define H1_DIM  1024
#define H2_DIM  512
#define C_DIM   64

#if defined(__gfx1250__) && __has_builtin(__builtin_amdgcn_global_load_async_to_lds_b128)
#define USE_ASYNC_LDS 1
#else
#define USE_ASYNC_LDS 0
#endif

#if USE_ASYNC_LDS
// probe-verified: param0 is v4i in addrspace(1)
typedef __attribute__((address_space(1))) v4i g_v4i;   // global
typedef __attribute__((address_space(3))) v4i l_v4i;   // LDS

__device__ __forceinline__ void async_cp_b128(const bf16_t* g, bf16_t* l) {
    __builtin_amdgcn_global_load_async_to_lds_b128((g_v4i*)g, (l_v4i*)l, 0, 0);
}
__device__ __forceinline__ void wait_async0() {
#if __has_builtin(__builtin_amdgcn_s_wait_asynccnt)
    __builtin_amdgcn_s_wait_asynccnt(0);
#else
    asm volatile("s_wait_asynccnt 0" ::: "memory");
#endif
}
#endif

// ---------------- f32 -> bf16 convert ----------------
__global__ void cvt_bf16_kernel(const float* __restrict__ in, bf16_t* __restrict__ out, int n) {
    int i = blockIdx.x * blockDim.x + threadIdx.x;
    if (i < n) out[i] = (bf16_t)in[i];
}

// ---------------- bf16 GEMM: C[M,N] = A[M,K] @ B[N,K]^T (+epilogue) ----------
// grouped: B/bias/outH indexed by blockIdx.z, epilogue scales acc by pmat[m,z].
template<int BM, int BN, int WAVES_M, int WAVES_N>
__global__ __launch_bounds__(32 * WAVES_M * WAVES_N)
void gemm_bf16_kernel(const bf16_t* __restrict__ A, const bf16_t* __restrict__ B,
                      int M, int N, int K,
                      const float* __restrict__ bias,
                      const float* __restrict__ pmat, int pC,
                      float* __restrict__ outF, bf16_t* __restrict__ outH,
                      int grouped)
{
    constexpr int NT  = 32 * WAVES_M * WAVES_N;  // threads per block
    constexpr int WM  = BM / WAVES_M;            // wave tile M
    constexpr int WN  = BN / WAVES_N;            // wave tile N
    constexpr int FM  = WM / 16;                 // 16x16 frags per wave (M)
    constexpr int FN  = WN / 16;
    constexpr int LDT = 40;                      // LDS row stride (bf16): 64B + 16B pad
    constexpr int CA  = (BM * 32) / (NT * 8);    // 16B chunks / thread (A)
    constexpr int CB  = (BN * 32) / (NT * 8);

    __shared__ bf16_t As[2][BM * LDT];
    __shared__ bf16_t Bs[2][BN * LDT];

    const int tid = threadIdx.x;
    const int z   = blockIdx.z;
    const bf16_t* Bb    = B    + (size_t)z * (size_t)N * (size_t)K;
    const float*  biasb = bias + (grouped ? (size_t)z * N : 0);

    const int m0 = blockIdx.y * BM;
    const int n0 = blockIdx.x * BN;

    const int lane = tid & 31;
    const int wave = tid >> 5;
    const int wm   = (wave / WAVES_N) * WM;
    const int wn   = (wave % WAVES_N) * WN;
    const int half = lane >> 4;                  // lane group (0: 0-15, 1: 16-31)
    const int ln   = lane & 15;

    v8f acc[FM][FN] = {};
    const int nsteps = K >> 5;

#if USE_ASYNC_LDS
    // ---- async global->LDS double buffering (no staging VGPRs) ----
    auto issue = [&](int buf, int k0) {
        #pragma unroll
        for (int i = 0; i < CA; ++i) {
            int c = tid + i * NT;
            async_cp_b128(A + (size_t)(m0 + (c >> 2)) * K + k0 + (c & 3) * 8,
                          &As[buf][(c >> 2) * LDT + (c & 3) * 8]);
        }
        #pragma unroll
        for (int i = 0; i < CB; ++i) {
            int c = tid + i * NT;
            async_cp_b128(Bb + (size_t)(n0 + (c >> 2)) * K + k0 + (c & 3) * 8,
                          &Bs[buf][(c >> 2) * LDT + (c & 3) * 8]);
        }
    };
    issue(0, 0);
    wait_async0();
    __syncthreads();
#else
    // ---- register-staged double buffering ----
    uint4 ra[CA], rb[CB];
    auto gload = [&](int k0) {
        #pragma unroll
        for (int i = 0; i < CA; ++i) {
            int c = tid + i * NT;
            ra[i] = *(const uint4*)(A + (size_t)(m0 + (c >> 2)) * K + k0 + (c & 3) * 8);
        }
        #pragma unroll
        for (int i = 0; i < CB; ++i) {
            int c = tid + i * NT;
            rb[i] = *(const uint4*)(Bb + (size_t)(n0 + (c >> 2)) * K + k0 + (c & 3) * 8);
        }
    };
    auto sstore = [&](int buf) {
        #pragma unroll
        for (int i = 0; i < CA; ++i) {
            int c = tid + i * NT;
            *(uint4*)&As[buf][(c >> 2) * LDT + (c & 3) * 8] = ra[i];
        }
        #pragma unroll
        for (int i = 0; i < CB; ++i) {
            int c = tid + i * NT;
            *(uint4*)&Bs[buf][(c >> 2) * LDT + (c & 3) * 8] = rb[i];
        }
    };
    gload(0);
    sstore(0);
    __syncthreads();
#endif

    for (int s = 0; s < nsteps; ++s) {
        const int buf = s & 1;
#if USE_ASYNC_LDS
        if (s + 1 < nsteps) issue(buf ^ 1, (s + 1) << 5);  // overlap with WMMA burst
#else
        if (s + 1 < nsteps) gload((s + 1) << 5);
#endif

        // A-frag (16x32 16-bit): lane ln = row M; half selects K-octets.
        v16bf af[FM], bfr[FN];
        #pragma unroll
        for (int i = 0; i < FM; ++i) {
            int row = wm + i * 16 + ln;
            union { v16bf v; v8bf h[2]; } u;
            u.h[0] = *(const v8bf*)&As[buf][row * LDT + half * 8];
            u.h[1] = *(const v8bf*)&As[buf][row * LDT + 16 + half * 8];
            af[i] = u.v;
        }
        // B-frag (32x16): lane ln = col N; half selects K range. B[k][n] == W[n][k].
        #pragma unroll
        for (int j = 0; j < FN; ++j) {
            int row = wn + j * 16 + ln;
            union { v16bf v; v8bf h[2]; } u;
            u.h[0] = *(const v8bf*)&Bs[buf][row * LDT + half * 8];
            u.h[1] = *(const v8bf*)&Bs[buf][row * LDT + 16 + half * 8];
            bfr[j] = u.v;
        }
        #pragma unroll
        for (int i = 0; i < FM; ++i)
            #pragma unroll
            for (int j = 0; j < FN; ++j)
                acc[i][j] = __builtin_amdgcn_wmma_f32_16x16x32_bf16(
                    false, af[i], false, bfr[j], (short)0, acc[i][j], false, false);

        if (s + 1 < nsteps) {
#if USE_ASYNC_LDS
            wait_async0();     // this wave's copies into buf^1 landed
            __syncthreads();   // all waves' copies visible
#else
            sstore(buf ^ 1);
            __syncthreads();
#endif
        }
    }

    // Epilogue. C/D layout: VGPR r -> M = half*8 + r, N = ln.
    float bv[FN];
    #pragma unroll
    for (int j = 0; j < FN; ++j) bv[j] = biasb[n0 + wn + j * 16 + ln];

    #pragma unroll
    for (int i = 0; i < FM; ++i) {
        const int rbase = m0 + wm + i * 16 + half * 8;
        float pv[8];
        if (grouped) {
            #pragma unroll
            for (int r = 0; r < 8; ++r) pv[r] = pmat[(size_t)(rbase + r) * pC + z];
        }
        #pragma unroll
        for (int j = 0; j < FN; ++j) {
            const int gc = n0 + wn + j * 16 + ln;
            #pragma unroll
            for (int r = 0; r < 8; ++r) {
                float v = acc[i][j][r];
                if (grouped) v *= pv[r];
                v += bv[j];
                if (outF) outF[(size_t)(rbase + r) * N + gc] = v;
                if (outH) outH[((size_t)z * M + rbase + r) * N + gc] = (bf16_t)v;
            }
        }
    }
}

// ------------- column-wise (axis=0) mean/var: partial sums via atomics -------
__global__ void col_partial_kernel(const float* __restrict__ Y, float* __restrict__ sum,
                                   float* __restrict__ ssq, int N, int rowsPerChunk) {
    int col = blockIdx.x * blockDim.x + threadIdx.x;
    int r0  = blockIdx.y * rowsPerChunk;
    float s = 0.f, q = 0.f;
    for (int r = r0; r < r0 + rowsPerChunk; ++r) {
        float v = Y[(size_t)r * N + col];
        s += v; q += v * v;
    }
    atomicAdd(&sum[col], s);
    atomicAdd(&ssq[col], q);
}

__global__ void col_finalize_kernel(float* __restrict__ sum, float* __restrict__ ssq,
                                    int M, int N, float eps) {
    int col = blockIdx.x * blockDim.x + threadIdx.x;
    if (col >= N) return;
    float mu  = sum[col] / (float)M;
    float var = ssq[col] / (float)M - mu * mu;
    sum[col] = mu;                 // reuse: sum -> mean
    ssq[col] = rsqrtf(var + eps);  // reuse: ssq -> rstd
}

// ------------- BN (precomputed mean/rstd) + LeakyReLU, emit bf16 -------------
__global__ void bn_act_kernel(const float* __restrict__ Y, const float* __restrict__ mean,
                              const float* __restrict__ rstd, const float* __restrict__ g,
                              const float* __restrict__ b, float slope,
                              bf16_t* __restrict__ out, int total, int nmask) {
    int i = blockIdx.x * blockDim.x + threadIdx.x;
    if (i >= total) return;
    int col = i & nmask;
    float v = g[col] * (Y[i] - mean[col]) * rstd[col] + b[col];
    v = (v >= 0.f) ? v : slope * v;
    out[i] = (bf16_t)v;
}

// ------------- row softmax over 64 classes ----------------------------------
__global__ void softmax64_kernel(const float* __restrict__ logits, float* __restrict__ p) {
    __shared__ float red[64];
    int b = blockIdx.x, t = threadIdx.x;
    float v = logits[(size_t)b * 64 + t];
    red[t] = v; __syncthreads();
    for (int s = 32; s > 0; s >>= 1) { if (t < s) red[t] = fmaxf(red[t], red[t + s]); __syncthreads(); }
    float m = red[0]; __syncthreads();
    float e = expf(v - m);
    red[t] = e; __syncthreads();
    for (int s = 32; s > 0; s >>= 1) { if (t < s) red[t] += red[t + s]; __syncthreads(); }
    p[(size_t)b * 64 + t] = e / red[0];
}

// ------------- per-(class, feature) batch stats over hd ---------------------
__global__ void hd_stats_kernel(const bf16_t* __restrict__ hd, float* __restrict__ mean,
                                float* __restrict__ rstd, int M, int N, float eps) {
    int c = blockIdx.y;
    int n = blockIdx.x * blockDim.x + threadIdx.x;
    const bf16_t* base = hd + (size_t)c * M * N + n;
    float s = 0.f, q = 0.f;
    for (int m = 0; m < M; ++m) {
        float v = (float)base[(size_t)m * N];
        s += v; q += v * v;
    }
    float mu  = s / (float)M;
    float var = q / (float)M - mu * mu;
    mean[(size_t)c * N + n] = mu;
    rstd[(size_t)c * N + n] = rsqrtf(var + eps);
}

// ------------- d[c,b] = sum_o Wd2[c,o]*lrelu(BN(hd)) + bd2[c] ---------------
__global__ __launch_bounds__(256)
void final_d_kernel(const bf16_t* __restrict__ hd, const float* __restrict__ mean,
                    const float* __restrict__ rstd, const float* __restrict__ gd,
                    const float* __restrict__ bdn, const float* __restrict__ Wd2,
                    const float* __restrict__ bd2, float* __restrict__ dout,
                    int M, int N) {
    int gw   = (blockIdx.x * blockDim.x + threadIdx.x) >> 5;
    int lane = threadIdx.x & 31;
    int c = gw / M;
    int b = gw - c * M;
    const bf16_t* row = hd + ((size_t)c * M + b) * N;
    const float* mu = mean + (size_t)c * N;
    const float* rs = rstd + (size_t)c * N;
    const float* gg = gd   + (size_t)c * N;
    const float* bb = bdn  + (size_t)c * N;
    const float* w2 = Wd2  + (size_t)c * N;
    float s = 0.f;
    for (int o = lane; o < N; o += 32) {
        float hv = (float)row[o];
        float t = gg[o] * (hv - mu[o]) * rs[o] + bb[o];
        t = (t >= 0.f) ? t : 0.2f * t;
        s += t * w2[o];
    }
    for (int off = 16; off > 0; off >>= 1) s += __shfl_down(s, off, 32);
    if (lane == 0) dout[(size_t)c * M + b] = s + bd2[c];
}

// ---------------------------------------------------------------------------
extern "C" void kernel_launch(void* const* d_in, const int* in_sizes, int n_in,
                              void* d_out, int out_size, void* d_ws, size_t ws_size,
                              hipStream_t stream) {
    const float* x   = (const float*)d_in[0];
    const float* We1 = (const float*)d_in[1];
    const float* be1 = (const float*)d_in[2];
    const float* g1  = (const float*)d_in[3];
    const float* b1  = (const float*)d_in[4];
    const float* We2 = (const float*)d_in[5];
    const float* be2 = (const float*)d_in[6];
    const float* g2  = (const float*)d_in[7];
    const float* b2  = (const float*)d_in[8];
    const float* Wc  = (const float*)d_in[9];
    const float* bc  = (const float*)d_in[10];
    const float* Wd1 = (const float*)d_in[11];
    const float* bd1 = (const float*)d_in[12];
    const float* gd  = (const float*)d_in[13];
    const float* bdn = (const float*)d_in[14];
    const float* Wd2 = (const float*)d_in[15];
    const float* bd2 = (const float*)d_in[16];

    float* c_out  = (float*)d_out;                          // [4096, 64]
    float* d_outp = c_out + (size_t)B_DIM * C_DIM;          // [64, 4096]

    // ---- workspace layout (256B aligned) ----
    char* base = (char*)d_ws;
    size_t off = 0;
    auto take = [&](size_t bytes) { char* p = base + off; off = (off + bytes + 255) & ~(size_t)255; return p; };
    bf16_t* xb   = (bf16_t*)take((size_t)B_DIM * DIN * 2);
    bf16_t* w1b  = (bf16_t*)take((size_t)H1_DIM * DIN * 2);
    bf16_t* hb   = (bf16_t*)take((size_t)B_DIM * H1_DIM * 2);
    bf16_t* w2b  = (bf16_t*)take((size_t)H2_DIM * H1_DIM * 2);
    bf16_t* fb   = (bf16_t*)take((size_t)B_DIM * H2_DIM * 2);
    bf16_t* wcb  = (bf16_t*)take((size_t)C_DIM * H2_DIM * 2);
    bf16_t* wd1b = (bf16_t*)take((size_t)C_DIM * H2_DIM * H2_DIM * 2);
    float*  Y1   = (float*)take((size_t)B_DIM * H1_DIM * 4);
    float*  Y2   = (float*)take((size_t)B_DIM * H2_DIM * 4);
    float*  p    = (float*)take((size_t)B_DIM * C_DIM * 4);
    float*  s1   = (float*)take((size_t)H1_DIM * 4);   // sum1 -> mean1
    float*  q1   = (float*)take((size_t)H1_DIM * 4);   // ssq1 -> rstd1
    float*  s2   = (float*)take((size_t)H2_DIM * 4);
    float*  q2   = (float*)take((size_t)H2_DIM * 4);
    float*  hmu  = (float*)take((size_t)C_DIM * H2_DIM * 4);
    float*  hrs  = (float*)take((size_t)C_DIM * H2_DIM * 4);
    bf16_t* hdb  = (bf16_t*)take((size_t)C_DIM * B_DIM * H2_DIM * 2);

    // ---- convert GEMM operands to bf16 ----
    auto cvt = [&](const float* src, bf16_t* dst, int n) {
        cvt_bf16_kernel<<<(n + 255) / 256, 256, 0, stream>>>(src, dst, n);
    };
    cvt(x,   xb,   B_DIM * DIN);
    cvt(We1, w1b,  H1_DIM * DIN);
    cvt(We2, w2b,  H2_DIM * H1_DIM);
    cvt(Wc,  wcb,  C_DIM * H2_DIM);
    cvt(Wd1, wd1b, C_DIM * H2_DIM * H2_DIM);

    // ---- GEMM1: Y1 = x @ We1^T + be1  [4096,1024] ----
    gemm_bf16_kernel<128,128,2,2><<<dim3(H1_DIM / 128, B_DIM / 128, 1), 128, 0, stream>>>(
        xb, w1b, B_DIM, H1_DIM, DIN, be1, nullptr, 0, Y1, nullptr, 0);

    // ---- BN1 stats (eps=0.8) + act -> h bf16 ----
    (void)hipMemsetAsync(s1, 0, (size_t)H1_DIM * 4, stream);
    (void)hipMemsetAsync(q1, 0, (size_t)H1_DIM * 4, stream);
    col_partial_kernel<<<dim3(H1_DIM / 256, 32), 256, 0, stream>>>(Y1, s1, q1, H1_DIM, B_DIM / 32);
    col_finalize_kernel<<<(H1_DIM + 255) / 256, 256, 0, stream>>>(s1, q1, B_DIM, H1_DIM, 0.8f);
    bn_act_kernel<<<(B_DIM * H1_DIM + 255) / 256, 256, 0, stream>>>(
        Y1, s1, q1, g1, b1, 0.01f, hb, B_DIM * H1_DIM, H1_DIM - 1);

    // ---- GEMM2: Y2 = h @ We2^T + be2  [4096,512] ----
    gemm_bf16_kernel<128,128,2,2><<<dim3(H2_DIM / 128, B_DIM / 128, 1), 128, 0, stream>>>(
        hb, w2b, B_DIM, H2_DIM, H1_DIM, be2, nullptr, 0, Y2, nullptr, 0);

    // ---- BN2 stats (eps=1e-5) + act -> f bf16 ----
    (void)hipMemsetAsync(s2, 0, (size_t)H2_DIM * 4, stream);
    (void)hipMemsetAsync(q2, 0, (size_t)H2_DIM * 4, stream);
    col_partial_kernel<<<dim3(H2_DIM / 256, 32), 256, 0, stream>>>(Y2, s2, q2, H2_DIM, B_DIM / 32);
    col_finalize_kernel<<<(H2_DIM + 255) / 256, 256, 0, stream>>>(s2, q2, B_DIM, H2_DIM, 1e-5f);
    bn_act_kernel<<<(B_DIM * H2_DIM + 255) / 256, 256, 0, stream>>>(
        Y2, s2, q2, g2, b2, 0.01f, fb, B_DIM * H2_DIM, H2_DIM - 1);

    // ---- GEMM3: logits c = f @ Wc^T + bc -> d_out [4096,64] ----
    gemm_bf16_kernel<64,64,2,2><<<dim3(C_DIM / 64, B_DIM / 64, 1), 128, 0, stream>>>(
        fb, wcb, B_DIM, C_DIM, H2_DIM, bc, nullptr, 0, c_out, nullptr, 0);

    // ---- softmax over classes ----
    softmax64_kernel<<<B_DIM, 64, 0, stream>>>(c_out, p);

    // ---- grouped GEMM: hd[c] = diag(p[:,c]) * (f @ Wd1[c]^T) + bd1[c] ------
    gemm_bf16_kernel<128,128,2,2><<<dim3(H2_DIM / 128, B_DIM / 128, C_DIM), 128, 0, stream>>>(
        fb, wd1b, B_DIM, H2_DIM, H2_DIM, bd1, p, C_DIM, nullptr, hdb, 1);

    // ---- per-(c,o) batch stats over hd ----
    hd_stats_kernel<<<dim3(H2_DIM / 128, C_DIM), 128, 0, stream>>>(
        hdb, hmu, hrs, B_DIM, H2_DIM, 1e-5f);

    // ---- d[c,b] = Wd2[c] . lrelu(BN(hd[c,b,:]), 0.2) + bd2[c] ----
    final_d_kernel<<<(C_DIM * B_DIM) / 8, 256, 0, stream>>>(
        hdb, hmu, hrs, gd, bdn, Wd2, bd2, d_outp, B_DIM, H2_DIM);
}
